// DeformableConv2D_36326833389674
// MI455X (gfx1250) — matrix operand order repaired and checked by reading, not compiled, not run on hardware
//
#include <hip/hip_runtime.h>
#include <math.h>

// ---------------------------------------------------------------------------
// DCNv2 (deformable conv v2) + BN + ReLU for MI455X (gfx1250, wave32, WMMA).
// Main GEMM (M=128, K=1152, N=32768, ~9.7 GFLOP) runs on v_wmma_f32_16x16x32_f16
// with an LDS-staged deformable im2col. x (16.8MB) lives in L2 (192MB), so the
// kernel is matrix-pipe bound, not HBM bound.
// ---------------------------------------------------------------------------

typedef __attribute__((ext_vector_type(16))) _Float16 v16h;
typedef __attribute__((ext_vector_type(8)))  _Float16 v8h;
typedef __attribute__((ext_vector_type(8)))  float    v8f;

#define BATCH 8
#define CIN   128
#define HH    64
#define WW    64
#define COUT  128
#define KK9   9
#define KTOT  1152            // CIN * KK9
#define NPIX  32768           // BATCH*HH*WW
#define KSTEPS 36             // KTOT / 32
#define MT     8              // COUT / 16
#define PPAD   1160           // patch row stride in halves: 2320B = 580 dwords == 4 mod 64 -> conflict-free

// workspace layout (float element offsets unless noted)
#define WS_POSY 0
#define WS_POSX 294912
#define WS_MOD  589824
#define WS_STAT 884736        // sum[128], sumsq[128]
#define WS_WHP_BYTES 3539968  // packed f16 weights, 32B aligned

// ---------------------------------------------------------------------------
__global__ void zero_stats(float* stats) {
    int t = blockIdx.x * blockDim.x + threadIdx.x;
    if (t < 256) stats[t] = 0.f;
}

// ---------------------------------------------------------------------------
// Repack w_reg (COUT,CIN,3,3) fp32 into f16 WMMA A fragments.
// Fragment linear index: ((mt*KSTEPS + s)*32 + lane)*16 + j
// A 16x32 f16 layout (ISA 7.12.2): lanes 0-15: M=lane, K = {0..7, 16..23};
//                                  lanes 16-31: M=lane-16, K = {8..15, 24..31}.
// K-index convention: Kg = kk*128 + c  (must match patch staging below).
__global__ void pack_w(const float* __restrict__ wreg, _Float16* __restrict__ whp) {
    int tid = blockIdx.x * 256 + threadIdx.x;
    if (tid >= MT * KSTEPS * 32 * 16) return;
    int j  = tid & 15;
    int l  = (tid >> 4) & 31;
    int s  = (tid >> 9) % KSTEPS;
    int mt = (tid >> 9) / KSTEPS;
    int M  = mt * 16 + (l & 15);
    int kmap = (l < 16) ? ((j < 8) ? j : j + 8)
                        : ((j < 8) ? j + 8 : j + 16);
    int Kg = s * 32 + kmap;
    int c  = Kg & 127;
    int kk = Kg >> 7;
    whp[tid] = (_Float16)wreg[(M * CIN + c) * KK9 + kk];
}

// ---------------------------------------------------------------------------
// Offset (18ch) + modulator (9ch) 3x3 convs, direct. One thread per pixel.
// Weights staged through LDS in 8-input-channel chunks (broadcast reads).
// Emits final sampling positions py/px and modulator m = 2*sigmoid(conv).
__global__ void offmod_kernel(const float* __restrict__ x,
                              const float* __restrict__ w_off,
                              const float* __restrict__ b_off,
                              const float* __restrict__ w_mod,
                              const float* __restrict__ b_mod,
                              float* __restrict__ ws) {
    __shared__ float wsh[27][8][9];   // 7776 B
    int t = threadIdx.x;
    int n = blockIdx.x * 256 + t;
    int b  = n >> 12;
    int hw = n & 4095;
    int h  = hw >> 6;
    int w  = hw & 63;
    const float* xb = x + (size_t)b * CIN * (HH * WW);

    float acc[27];
#pragma unroll
    for (int i = 0; i < 27; i++) acc[i] = 0.f;

    for (int cc = 0; cc < 16; ++cc) {
        __syncthreads();
        for (int i = t; i < 27 * 8 * 9; i += 256) {
            int oc = i / 72; int r = i % 72; int c8 = r / 9; int kk = r % 9;
            int c = cc * 8 + c8;
            wsh[oc][c8][kk] = (oc < 18) ? w_off[(oc * CIN + c) * 9 + kk]
                                        : w_mod[((oc - 18) * CIN + c) * 9 + kk];
        }
        __syncthreads();
#pragma unroll
        for (int c8 = 0; c8 < 8; c8++) {
            int c = cc * 8 + c8;
            const float* xc = xb + (size_t)c * (HH * WW);
            float tap[9];
#pragma unroll
            for (int kk = 0; kk < 9; kk++) {
                int yy = h - 1 + kk / 3;
                int xx = w - 1 + kk % 3;
                bool ok = (yy >= 0) && (yy < HH) && (xx >= 0) && (xx < WW);
                tap[kk] = ok ? xc[yy * WW + xx] : 0.f;
            }
#pragma unroll
            for (int oc = 0; oc < 27; oc++) {
                float s = 0.f;
#pragma unroll
                for (int kk = 0; kk < 9; kk++) s += wsh[oc][c8][kk] * tap[kk];
                acc[oc] += s;
            }
        }
    }

    float* posy = ws + WS_POSY;
    float* posx = ws + WS_POSX;
    float* modp = ws + WS_MOD;
#pragma unroll
    for (int kk = 0; kk < 9; kk++) {
        float dy = acc[kk * 2 + 0] + b_off[kk * 2 + 0];
        float dx = acc[kk * 2 + 1] + b_off[kk * 2 + 1];
        float mv = acc[18 + kk]    + b_mod[kk];
        float m  = 2.f / (1.f + __expf(-mv));
        int ki = kk / 3, kj = kk % 3;
        int idx = ((b * 9 + kk) * HH + h) * WW + w;
        posy[idx] = (float)(h - 1 + ki) + dy;
        posx[idx] = (float)(w - 1 + kj) + dx;
        modp[idx] = m;
    }
}

// ---------------------------------------------------------------------------
// Deformable im2col (LDS, f16) + WMMA GEMM. Block = 256 threads = 8 waves.
// Tile: 16 output pixels x 128 output channels; wave w owns one 16x16 M-tile.
__global__ void deform_gemm(const float* __restrict__ x,
                            const _Float16* __restrict__ whp,
                            const float* __restrict__ ws,
                            const float* __restrict__ b_reg,
                            float* __restrict__ out,
                            float* __restrict__ stats) {
    __shared__ __align__(32) _Float16 patch[16][PPAD];   // 37,120 B

    int t  = threadIdx.x;
    int n0 = blockIdx.x * 16;
    int b  = n0 >> 12;
    int h  = (n0 & 4095) >> 6;
    int w0 = n0 & 63;                 // multiple of 16

    const float* posy = ws + WS_POSY;
    const float* posx = ws + WS_POSX;
    const float* modp = ws + WS_MOD;

    // ---- stage deformable patch: 16 pixels x (kk,c) -> f16 ----
    int g  = t >> 4;                  // pixel in tile 0..15
    int c0 = (t & 15) * 8;            // 8 input channels per thread
    int w  = w0 + g;
    for (int kk = 0; kk < 9; kk++) {
        int pidx = ((b * 9 + kk) * HH + h) * WW + w;
        float py = posy[pidx], px = posx[pidx], m = modp[pidx];
        float y0f = floorf(py), x0f = floorf(px);
        float ly = py - y0f, lx = px - x0f;
        int y0 = (int)y0f, x0 = (int)x0f;
        int y1 = y0 + 1,   x1 = x0 + 1;
        float vy0 = (y0 >= 0 && y0 < HH) ? 1.f : 0.f;
        float vy1 = (y1 >= 0 && y1 < HH) ? 1.f : 0.f;
        float vx0 = (x0 >= 0 && x0 < WW) ? 1.f : 0.f;
        float vx1 = (x1 >= 0 && x1 < WW) ? 1.f : 0.f;
        float w00 = (1.f - ly) * (1.f - lx) * vy0 * vx0;
        float w01 = (1.f - ly) * lx         * vy0 * vx1;
        float w10 = ly * (1.f - lx)         * vy1 * vx0;
        float w11 = ly * lx                 * vy1 * vx1;
        int y0c = min(max(y0, 0), HH - 1), y1c = min(max(y1, 0), HH - 1);
        int x0c = min(max(x0, 0), WW - 1), x1c = min(max(x1, 0), WW - 1);
        int o00 = y0c * WW + x0c, o01 = y0c * WW + x1c;
        int o10 = y1c * WW + x0c, o11 = y1c * WW + x1c;
        const float* xb = x + ((size_t)b * CIN + c0) * (HH * WW);
#pragma unroll
        for (int cj = 0; cj < 8; cj++) {
            const float* xc = xb + (size_t)cj * (HH * WW);
            float v = w00 * xc[o00] + w01 * xc[o01] + w10 * xc[o10] + w11 * xc[o11];
            patch[g][kk * 128 + c0 + cj] = (_Float16)(v * m);
        }
    }
    __syncthreads();

    // ---- WMMA GEMM: 36 steps of 16x16x32 ----
    int wv = t >> 5;                  // M-tile 0..7
    int l  = t & 31;
    int nIdx  = l & 15;               // B-frag column (pixel)
    int koffB = (l >= 16) ? 8 : 0;    // B 32x16 f16 layout: K = {0..7,16..23} / {8..15,24..31}
    const _Float16* aBase = whp + (size_t)wv * (KSTEPS * 512) + (size_t)l * 16;

    v8f acc = {};
#pragma unroll 4
    for (int s = 0; s < KSTEPS; s++) {
        v16h a = *(const v16h*)(aBase + (size_t)s * 512);        // 32B coalesced
        const _Float16* bp = &patch[nIdx][s * 32 + koffB];
        v8h b0 = *(const v8h*)(bp);
        v8h b1 = *(const v8h*)(bp + 16);
        v16h bb = __builtin_shufflevector(b0, b1,
                    0, 1, 2, 3, 4, 5, 6, 7, 8, 9, 10, 11, 12, 13, 14, 15);
        acc = __builtin_amdgcn_wmma_f32_16x16x32_f16(
                  false, a, false, bb, (short)0, acc, false, false);
    }

    // ---- epilogue: bias, store y, per-channel sum/sumsq for BN ----
    int mhalf = (l >= 16) ? 8 : 0;    // C/D layout: lanes 16-31 hold M = r+8
    int pix   = l & 15;
    size_t outBase = (size_t)b * COUT * (HH * WW) + (size_t)h * WW + (w0 + pix);
#pragma unroll
    for (int r = 0; r < 8; r++) {
        int oc = wv * 16 + r + mhalf;
        float y = acc[r] + b_reg[oc];
        out[outBase + (size_t)oc * (HH * WW)] = y;
        float s1 = y, s2 = y * y;
#pragma unroll
        for (int msk = 1; msk < 16; msk <<= 1) {   // reduce over the 16 pixels sharing oc
            s1 += __shfl_xor(s1, msk, 32);
            s2 += __shfl_xor(s2, msk, 32);
        }
        if (pix == 0) {
            atomicAdd(&stats[oc], s1);
            atomicAdd(&stats[128 + oc], s2);
        }
    }
}

// ---------------------------------------------------------------------------
__global__ void bn_relu(float* __restrict__ out,
                        const float* __restrict__ stats,
                        const float* __restrict__ gamma,
                        const float* __restrict__ beta) {
    int i = blockIdx.x * 256 + threadIdx.x;      // float4 index
    if (i >= (NPIX * COUT) / 4) return;
    int c = (i >> 10) & 127;                     // ((i*4)/4096) % 128
    float inv = 1.f / 32768.f;
    float mean = stats[c] * inv;
    float var  = stats[128 + c] * inv - mean * mean;
    float sc = gamma[c] * rsqrtf(var + 1e-5f);
    float sh = beta[c] - mean * sc;
    float4 v = ((float4*)out)[i];
    v.x = fmaxf(v.x * sc + sh, 0.f);
    v.y = fmaxf(v.y * sc + sh, 0.f);
    v.z = fmaxf(v.z * sc + sh, 0.f);
    v.w = fmaxf(v.w * sc + sh, 0.f);
    ((float4*)out)[i] = v;
}

// ---------------------------------------------------------------------------
extern "C" void kernel_launch(void* const* d_in, const int* in_sizes, int n_in,
                              void* d_out, int out_size, void* d_ws, size_t ws_size,
                              hipStream_t stream) {
    const float* x     = (const float*)d_in[0];
    const float* w_off = (const float*)d_in[1];
    const float* b_off = (const float*)d_in[2];
    const float* w_mod = (const float*)d_in[3];
    const float* b_mod = (const float*)d_in[4];
    const float* w_reg = (const float*)d_in[5];
    const float* b_reg = (const float*)d_in[6];
    const float* gamma = (const float*)d_in[7];
    const float* beta  = (const float*)d_in[8];

    float*     out   = (float*)d_out;
    float*     ws    = (float*)d_ws;
    float*     stats = ws + WS_STAT;
    _Float16*  whp   = (_Float16*)((char*)d_ws + WS_WHP_BYTES);

    zero_stats<<<dim3(1), dim3(256), 0, stream>>>(stats);
    pack_w<<<dim3(576), dim3(256), 0, stream>>>(w_reg, whp);
    offmod_kernel<<<dim3(NPIX / 256), dim3(256), 0, stream>>>(x, w_off, b_off, w_mod, b_mod, ws);
    deform_gemm<<<dim3(NPIX / 16), dim3(256), 0, stream>>>(x, whp, ws, b_reg, out, stats);
    bn_relu<<<dim3((NPIX * COUT) / 4 / 256), dim3(256), 0, stream>>>(out, stats, gamma, beta);
}
